// SCEncoder_42872363548918
// MI455X (gfx1250) — matrix operand order repaired and checked by reading, not compiled, not run on hardware
//
#include <hip/hip_runtime.h>
#include <hip/hip_bf16.h>

// ---------- problem constants (from reference) ----------
#define NPOS 256   // N
#define KINF 128   // K
#define BB   256   // batch
#define DIM  256   // D
#define HID  512   // H

typedef __attribute__((ext_vector_type(16))) __bf16 v16bf;
typedef __attribute__((ext_vector_type(8)))  float  v8f;
typedef __attribute__((ext_vector_type(4)))  unsigned int u32x4;
typedef int i4v __attribute__((vector_size(4 * sizeof(int))));   // matches builtin param

union FragU { v16bf v; u32x4 q[2]; };

#define GAS __attribute__((address_space(1)))
#define LAS __attribute__((address_space(3)))

#if defined(__gfx1250__) && __has_builtin(__builtin_amdgcn_global_load_async_to_lds_b128)
#define USE_ASYNC 1
#else
#define USE_ASYNC 0
#endif

__device__ __forceinline__ void wait_async0()
{
#if defined(__gfx1250__) && __has_builtin(__builtin_amdgcn_s_wait_asynccnt)
    __builtin_amdgcn_s_wait_asynccnt(0);
#else
    asm volatile("s_wait_asynccnt 0" ::: "memory");
#endif
}

// =====================================================================
// Fused gather + GEMM:  Out[M,Nout] = act( A[M,Kin] @ Wt^T + bias )
//   AMODE 0: A = plain bf16 [M,Kin]
//   AMODE 1: A = concat(e_odd, e_even)            (cn input, Kin=2D)
//   AMODE 2: A = concat(e_odd, e_even, label[h1]) (bn input, Kin=3D)
//   Ebuf == nullptr (root) -> e rows are obs_emb[2]
// Wt is bf16 [Nout][Kin] (pre-transposed).
// Block tile 64x64, K-step 64, 8 waves (2x4), each wave computes 32x16
// => 4 v_wmma_f32_16x16x32_bf16 per wave per staging round.
// Staging: async global->LDS (ASYNCcnt) with double-buffered LDS when
// available; else register-prefetch + ds_store.
// =====================================================================
template<int AMODE, bool RELU>
__global__ __launch_bounds__(256)
void gemm_ws(const __bf16* __restrict__ A,
             const __bf16* __restrict__ Ebuf,
             const int*    __restrict__ H1,
             const __bf16* __restrict__ obs_bf,
             const __bf16* __restrict__ lab_bf,
             int halfShift,
             const __bf16* __restrict__ Wt,
             const float*  __restrict__ bias,
             __bf16* __restrict__ Out,
             int M, int Kin, int Nout)
{
    // LDS tiles: row/col stride 72 bf16 (=144B) to kill bank conflicts,
    // 16B aligned for b128 access. A: rows x K (row-major), B: cols x K.
    __shared__ __align__(16) __bf16 As[2][64 * 72];
    __shared__ __align__(16) __bf16 Bs[2][64 * 72];

    const int t     = threadIdx.x;
    const int lane  = t & 31;
    const int wave  = t >> 5;
    const int waveM = wave >> 2;          // 0..1 -> 32-row strip
    const int waveN = wave & 3;           // 0..3 -> 16-col strip
    const int rowBase = blockIdx.y * 64;
    const int colBase = blockIdx.x * 64;

    // cooperative-load indices: 16 bf16 (32B) per thread per matrix
    const int a_r = t >> 2;               // 0..63  A tile row
    const int a_k = (t & 3) << 4;         // 0,16,32,48
    const int b_n = t & 63;               // 0..63  B tile col
    const int b_k = (t >> 6) << 4;        // 0,16,32,48

    const int gm = rowBase + a_r;
    int e_rowoff = 0, lab_sel = 0;
    if (AMODE >= 1) {
        const int half = 1 << halfShift;
        const int b = gm >> halfShift;
        const int j = gm & (half - 1);
        e_rowoff = (b * (half << 1) + (j << 1)) * DIM;   // &e[b, 2j, 0]
        if (AMODE == 2) lab_sel = H1[gm];                // hard bit 0/1
    }

    auto aSrc = [&](int k0) -> const __bf16* {
        const int gk = k0 + a_k;                 // 16-elem chunk within one segment
        if (AMODE == 0) return A + (size_t)gm * Kin + gk;
        const int s  = gk >> 8;                  // 0=e_odd 1=e_even 2=label
        const int kk = gk & (DIM - 1);
        if (AMODE == 2 && s == 2) return lab_bf + lab_sel * DIM + kk;
        if (Ebuf)                 return Ebuf + e_rowoff + s * DIM + kk;
        return obs_bf + 2 * DIM + kk;            // root: e == obs_emb[2]
    };
    auto bSrc = [&](int k0) -> const __bf16* {
        return Wt + (size_t)(colBase + b_n) * Kin + k0 + b_k;
    };

    v8f acc0 = {}, acc1 = {};
    const int g  = lane >> 4;
    const int lm = lane & 15;

    auto compute = [&](int buf) {
        const __bf16* as = As[buf];
        const __bf16* bs = Bs[buf];
#pragma unroll
        for (int kf = 0; kf < 2; ++kf) {
            FragU fb, fa0, fa1;
            // B: lane holds col lm, K = kf*32 + {g*8..g*8+7, 16+g*8..16+g*8+7}
            fb.q[0]  = *(const u32x4*)&bs[(waveN * 16 + lm) * 72 + kf * 32 + g * 8];
            fb.q[1]  = *(const u32x4*)&bs[(waveN * 16 + lm) * 72 + kf * 32 + 16 + g * 8];
            // A: lane holds row, K = kf*32 + {g*8..g*8+7, 16+g*8..16+g*8+7}
            fa0.q[0] = *(const u32x4*)&as[(waveM * 32 + lm) * 72 + kf * 32 + g * 8];
            fa0.q[1] = *(const u32x4*)&as[(waveM * 32 + lm) * 72 + kf * 32 + 16 + g * 8];
            fa1.q[0] = *(const u32x4*)&as[(waveM * 32 + 16 + lm) * 72 + kf * 32 + g * 8];
            fa1.q[1] = *(const u32x4*)&as[(waveM * 32 + 16 + lm) * 72 + kf * 32 + 16 + g * 8];
            acc0 = __builtin_amdgcn_wmma_f32_16x16x32_bf16(
                       false, fa0.v, false, fb.v, (short)0, acc0, false, false);
            acc1 = __builtin_amdgcn_wmma_f32_16x16x32_bf16(
                       false, fa1.v, false, fb.v, (short)0, acc1, false, false);
        }
    };

    const int nT = Kin >> 6;   // K-steps of 64

#if USE_ASYNC
    // -------- async global->LDS, double-buffered, 1 barrier / K-step ----
    auto stage = [&](int k0, int buf) {
        GAS i4v* sa = (GAS i4v*)aSrc(k0);
        GAS i4v* sb = (GAS i4v*)bSrc(k0);
        LAS i4v* da = (LAS i4v*)&As[buf][a_r * 72 + a_k];
        LAS i4v* db = (LAS i4v*)&Bs[buf][b_n * 72 + b_k];
        __builtin_amdgcn_global_load_async_to_lds_b128(sa, da, 0, 0);
        __builtin_amdgcn_global_load_async_to_lds_b128(sa, da, 16, 0);
        __builtin_amdgcn_global_load_async_to_lds_b128(sb, db, 0, 0);
        __builtin_amdgcn_global_load_async_to_lds_b128(sb, db, 16, 0);
    };
    stage(0, 0);
    for (int kt = 0; kt < nT; ++kt) {
        wait_async0();        // this wave's tile kt landed in LDS
        __syncthreads();      // everyone's tile kt landed; also: all waves are
                              // done reading the buffer stage() writes next
        if (kt + 1 < nT) stage((kt + 1) << 6, (kt + 1) & 1);
        compute(kt & 1);
    }
#else
    // -------- fallback: register prefetch + ds_store, 2 barriers / step --
    u32x4 ra0, ra1, rb0, rb1;
    auto ldregs = [&](int k0) {
        const u32x4* sa = (const u32x4*)aSrc(k0);
        const u32x4* sb = (const u32x4*)bSrc(k0);
        ra0 = sa[0]; ra1 = sa[1];
        rb0 = sb[0]; rb1 = sb[1];
    };
    ldregs(0);
    for (int kt = 0; kt < nT; ++kt) {
        *(u32x4*)&As[0][a_r * 72 + a_k]     = ra0;
        *(u32x4*)&As[0][a_r * 72 + a_k + 8] = ra1;
        *(u32x4*)&Bs[0][b_n * 72 + b_k]     = rb0;
        *(u32x4*)&Bs[0][b_n * 72 + b_k + 8] = rb1;
        __syncthreads();
        if (kt + 1 < nT) ldregs((kt + 1) << 6);   // overlap next tile with compute
        compute(0);
        __syncthreads();
    }
#endif

    // ---- epilogue: bias (+ReLU), store bf16 ----
    const int col = colBase + waveN * 16 + lm;
    const float bv = bias[col];
    const int r0 = rowBase + waveM * 32 + ((lane >> 4) << 3);
#pragma unroll
    for (int r = 0; r < 8; ++r) {
        float v0 = acc0[r] + bv;
        float v1 = acc1[r] + bv;
        if (RELU) { v0 = fmaxf(v0, 0.f); v1 = fmaxf(v1, 0.f); }
        Out[(size_t)(r0 + r) * Nout + col]      = (__bf16)v0;
        Out[(size_t)(r0 + 16 + r) * Nout + col] = (__bf16)v1;
    }
}

// =====================================================================
// Leaf (n==1): logits = e @ llr_W + llr_b ; softmax ; hard decision
// =====================================================================
__global__ void leaf_kernel(const __bf16* __restrict__ e,       // [B][D]
                            const float* __restrict__ llr_W,    // [D][2]
                            const float* __restrict__ llr_b,    // [2]
                            const int*   __restrict__ info_bits,// [B][K]
                            const float* __restrict__ rv,       // [B][N]
                            const int*   __restrict__ u_map,    // [N]
                            int off,
                            int*   __restrict__ dst_h,          // [B]
                            int*   __restrict__ u_buf,          // [B][N]
                            float* __restrict__ p_out)          // [B][N][2]
{
    const int b = blockIdx.x * blockDim.x + threadIdx.x;
    if (b >= BB) return;
    float l0 = llr_b[0], l1 = llr_b[1];
    for (int k = 0; k < DIM; ++k) {
        const float ev = (float)e[b * DIM + k];
        l0 += ev * llr_W[k * 2 + 0];
        l1 += ev * llr_W[k * 2 + 1];
    }
    const float m   = fmaxf(l0, l1);
    const float e0  = __expf(l0 - m), e1 = __expf(l1 - m);
    const float inv = 1.f / (e0 + e1);
    const float p0 = e0 * inv, p1 = e1 * inv;
    const int um   = u_map[off];
    const int fv   = (um >= 0) ? info_bits[b * KINF + um] : 2;
    const int hard = (rv[b * NPOS + off] > p0) ? 1 : 0;
    const int x    = (fv == 2) ? hard : fv;
    dst_h[b] = x;
    u_buf[b * NPOS + off] = x;
    p_out[(b * NPOS + off) * 2 + 0] = p0;
    p_out[(b * NPOS + off) * 2 + 1] = p1;
}

// dst[b,2j] = (h1+h2)&1 ; dst[b,2j+1] = h2   (polar re-encode)
__global__ void combine_kernel(const int* __restrict__ hl, const int* __restrict__ hr,
                               int* __restrict__ dst, int halfShift)
{
    const int half = 1 << halfShift;
    const int i = blockIdx.x * blockDim.x + threadIdx.x;
    if (i >= BB * half) return;
    const int b = i >> halfShift;
    const int j = i & (half - 1);
    const int h1 = hl[i], h2 = hr[i];
    dst[b * (half << 1) + (j << 1)]     = (h1 + h2) & 1;
    dst[b * (half << 1) + (j << 1) + 1] = h2;
}

__global__ void finalize_kernel(const int* __restrict__ h_root,
                                const int* __restrict__ u_buf,
                                const int* __restrict__ u_map,
                                float* __restrict__ out)  // x | f_out | u | (p_u by leaves)
{
    const int i = blockIdx.x * blockDim.x + threadIdx.x;
    if (i >= BB * NPOS) return;
    const int pos = i & (NPOS - 1);
    out[i] = (float)h_root[i];                                       // x
    out[BB * NPOS + i] = (u_map[pos] >= 0) ? 2.0f : (float)u_buf[i]; // f_out
    out[2 * BB * NPOS + i] = (float)u_buf[i];                        // u
}

// ---------- one-time prep: convert/transpose weights, build u_map ----------
__global__ void conv_transpose_kernel(const float* __restrict__ W, __bf16* __restrict__ Wt,
                                      int Kin, int Nout)
{
    const int i = blockIdx.x * blockDim.x + threadIdx.x;
    if (i >= Kin * Nout) return;
    const int k = i / Nout, n = i - k * Nout;
    Wt[(size_t)n * Kin + k] = (__bf16)W[i];
}
__global__ void conv_kernel(const float* __restrict__ s, __bf16* __restrict__ d, int n)
{
    const int i = blockIdx.x * blockDim.x + threadIdx.x;
    if (i < n) d[i] = (__bf16)s[i];
}
__global__ void umap_init_kernel(int* __restrict__ u_map)
{
    const int i = blockIdx.x * blockDim.x + threadIdx.x;
    if (i < NPOS) u_map[i] = -1;
}
__global__ void umap_set_kernel(const int* __restrict__ info_set, int* __restrict__ u_map)
{
    const int k = blockIdx.x * blockDim.x + threadIdx.x;
    if (k < KINF) u_map[info_set[k]] = k;
}

// =====================================================================
// Host-side DFS orchestration
// =====================================================================
struct Ctx {
    hipStream_t stream;
    const __bf16 *cnW1t, *cnW2t, *bnW1t, *bnW2t, *obs_bf, *lab_bf;
    const float  *cn_b1, *cn_b2, *bn_b1, *bn_b2, *llr_W, *llr_b, *rv;
    const int    *info_bits;
    int          *u_map, *u_buf, *h_root;
    int          *hl[9], *hr[9];
    __bf16       *e_buf[9];
    __bf16       *hidden;
    float        *p_out;
};

static void encode_node(Ctx& c, int depth, int off, int n, int* dst_h)
{
    if (n == 1) {
        leaf_kernel<<<1, 256, 0, c.stream>>>(c.e_buf[depth], c.llr_W, c.llr_b,
                                             c.info_bits, c.rv, c.u_map, off,
                                             dst_h, c.u_buf, c.p_out);
        return;
    }
    const int half = n >> 1;
    const int hs   = 7 - depth;            // log2(half)
    const int M    = BB * half;            // tokens (always multiple of 64)
    const __bf16* Eb = (depth == 0) ? nullptr : c.e_buf[depth];
    const dim3 blk(256);
    const dim3 g1(HID / 64, M / 64);
    const dim3 g2(DIM / 64, M / 64);

    // cn MLP -> left child e
    gemm_ws<1, true ><<<g1, blk, 0, c.stream>>>(nullptr, Eb, nullptr, c.obs_bf, c.lab_bf,
                                                hs, c.cnW1t, c.cn_b1, c.hidden, M, 2 * DIM, HID);
    gemm_ws<0, false><<<g2, blk, 0, c.stream>>>(c.hidden, nullptr, nullptr, nullptr, nullptr,
                                                0, c.cnW2t, c.cn_b2, c.e_buf[depth + 1], M, HID, DIM);
    encode_node(c, depth + 1, off, half, c.hl[depth + 1]);

    // bn MLP (uses left child hard bits) -> right child e
    gemm_ws<2, true ><<<g1, blk, 0, c.stream>>>(nullptr, Eb, c.hl[depth + 1], c.obs_bf, c.lab_bf,
                                                hs, c.bnW1t, c.bn_b1, c.hidden, M, 3 * DIM, HID);
    gemm_ws<0, false><<<g2, blk, 0, c.stream>>>(c.hidden, nullptr, nullptr, nullptr, nullptr,
                                                0, c.bnW2t, c.bn_b2, c.e_buf[depth + 1], M, HID, DIM);
    encode_node(c, depth + 1, off + half, half, c.hr[depth + 1]);

    combine_kernel<<<(BB * half + 255) / 256, blk, 0, c.stream>>>(
        c.hl[depth + 1], c.hr[depth + 1], dst_h, hs);
}

extern "C" void kernel_launch(void* const* d_in, const int* in_sizes, int n_in,
                              void* d_out, int out_size, void* d_ws, size_t ws_size,
                              hipStream_t stream)
{
    (void)in_sizes; (void)n_in; (void)out_size; (void)ws_size;
    // inputs in setup_inputs() order
    const int*   info_bits = (const int*)  d_in[0];
    const float* rv        = (const float*)d_in[1];
    const int*   info_set  = (const int*)  d_in[2];
    const float* obs_emb   = (const float*)d_in[3];
    const float* label_emb = (const float*)d_in[4];
    const float* cn_W1     = (const float*)d_in[5];
    const float* cn_b1     = (const float*)d_in[6];
    const float* cn_W2     = (const float*)d_in[7];
    const float* cn_b2     = (const float*)d_in[8];
    const float* bn_W1     = (const float*)d_in[9];
    const float* bn_b1     = (const float*)d_in[10];
    const float* bn_W2     = (const float*)d_in[11];
    const float* bn_b2     = (const float*)d_in[12];
    const float* llr_W     = (const float*)d_in[13];
    const float* llr_b     = (const float*)d_in[14];

    // carve workspace
    char* p = (char*)d_ws;
    auto carve = [&](size_t bytes) -> void* {
        void* r = (void*)p;
        p += (bytes + 255) & ~(size_t)255;
        return r;
    };
    __bf16* cnW1t  = (__bf16*)carve((size_t)2 * DIM * HID * 2);
    __bf16* cnW2t  = (__bf16*)carve((size_t)HID * DIM * 2);
    __bf16* bnW1t  = (__bf16*)carve((size_t)3 * DIM * HID * 2);
    __bf16* bnW2t  = (__bf16*)carve((size_t)HID * DIM * 2);
    __bf16* obs_bf = (__bf16*)carve((size_t)3 * DIM * 2);
    __bf16* lab_bf = (__bf16*)carve((size_t)3 * DIM * 2);
    int* u_map  = (int*)carve(NPOS * sizeof(int));
    int* h_root = (int*)carve((size_t)BB * NPOS * sizeof(int));
    int* u_buf  = (int*)carve((size_t)BB * NPOS * sizeof(int));

    Ctx c;
    c.stream = stream;
    c.cnW1t = cnW1t; c.cnW2t = cnW2t; c.bnW1t = bnW1t; c.bnW2t = bnW2t;
    c.obs_bf = obs_bf; c.lab_bf = lab_bf;
    c.cn_b1 = cn_b1; c.cn_b2 = cn_b2; c.bn_b1 = bn_b1; c.bn_b2 = bn_b2;
    c.llr_W = llr_W; c.llr_b = llr_b; c.rv = rv; c.info_bits = info_bits;
    c.u_map = u_map; c.u_buf = u_buf; c.h_root = h_root;
    c.e_buf[0] = nullptr; c.hl[0] = nullptr; c.hr[0] = nullptr;
    for (int d = 1; d <= 8; ++d) {
        const int n = NPOS >> d;
        c.hl[d]    = (int*)carve((size_t)BB * n * sizeof(int));
        c.hr[d]    = (int*)carve((size_t)BB * n * sizeof(int));
        c.e_buf[d] = (__bf16*)carve((size_t)BB * n * DIM * 2);
    }
    c.hidden = (__bf16*)carve((size_t)BB * (NPOS / 2) * HID * 2);
    c.p_out  = (float*)d_out + 3 * BB * NPOS;   // p_u section, written by leaves

    // ---- prep: weights -> bf16 transposed, embeddings -> bf16, u_map ----
    auto ct = [&](const float* W, __bf16* Wt, int Kin, int Nout) {
        conv_transpose_kernel<<<(Kin * Nout + 255) / 256, 256, 0, stream>>>(W, Wt, Kin, Nout);
    };
    ct(cn_W1, cnW1t, 2 * DIM, HID);
    ct(cn_W2, cnW2t, HID, DIM);
    ct(bn_W1, bnW1t, 3 * DIM, HID);
    ct(bn_W2, bnW2t, HID, DIM);
    conv_kernel<<<(3 * DIM + 255) / 256, 256, 0, stream>>>(obs_emb, obs_bf, 3 * DIM);
    conv_kernel<<<(3 * DIM + 255) / 256, 256, 0, stream>>>(label_emb, lab_bf, 3 * DIM);
    umap_init_kernel<<<1, 256, 0, stream>>>(u_map);
    umap_set_kernel<<<1, 128, 0, stream>>>(info_set, u_map);

    // ---- sequential SC tree (DFS, 511 nodes) ----
    encode_node(c, 0, 0, NPOS, h_root);

    // ---- outputs: x, f_out, u (p_u already written by leaves) ----
    finalize_kernel<<<(BB * NPOS + 255) / 256, 256, 0, stream>>>(h_root, u_buf, u_map,
                                                                 (float*)d_out);
}
